// VLMo_15418932593234
// MI455X (gfx1250) — matrix operand order repaired and checked by reading, not compiled, not run on hardware
//
#include <hip/hip_runtime.h>
#include <hip/hip_bf16.h>

#define DIM 768
#define HEADS 12
#define HEAD_DIM 64
#define HIDDEN 3072
#define BB 16
#define NN 236
#define TEXT_LEN 40
#define IMAG_LEN 196
#define ROWS (BB * NN)           // 3776
#define SCALE 0.125f             // 64^-0.5

typedef __attribute__((ext_vector_type(16))) _Float16 v16h;
typedef __attribute__((ext_vector_type(8)))  float    v8f;

union AF { v16h v; unsigned u[8]; };

// ---- WMMA fragment helpers (layouts per CDNA5 ISA 7.12.2, wave32) ----

// A fragment (16x32 f16, MxK): lane holds row M=lane&15.
// u32 vv: vv<4 -> K = 2*vv + (lane&16 ? 8:0); vv>=4 -> K = 16 + 2*(vv-4) + (lane&16 ? 8:0)
__device__ inline v16h frag_a(const _Float16* rowp, int kb, int lane) {
    AF f;
    const int hi = (lane & 16) >> 1;  // +8 for upper half-wave
#pragma unroll
    for (int vv = 0; vv < 8; ++vv) {
        int k0 = kb + ((vv & 4) << 2) + hi + ((vv & 3) << 1);
        f.u[vv] = *(const unsigned*)(rowp + k0);
    }
    return f.v;
}

// B fragment (32x16 f16, KxN): lane holds column N=lane&15, K order same as A fragment.
// colp points at &B[n][0] with K contiguous (i.e. [N][K] storage).
__device__ inline v16h frag_b(const _Float16* colp, int kb, int lane) {
    AF f;
    const int hi = (lane & 16) >> 1;
#pragma unroll
    for (int vv = 0; vv < 8; ++vv)
        f.u[vv] = *(const unsigned*)(colp + kb + ((vv & 4) << 2) + hi + ((vv & 3) << 1));
    return f.v;
}

__device__ inline v8f wmma16(v16h a, v16h b, v8f c) {
    return __builtin_amdgcn_wmma_f32_16x16x32_f16(false, a, false, b, (short)0, c, false, false);
}

// ---- async direct-to-LDS copy (GLOBAL_LOAD_ASYNC_TO_LDS_B128, ASYNCcnt) ----
// Copies 64 bytes per lane (4 x b128). dst low 32 bits of generic pointer = LDS address
// src is a 64-bit global address in a VGPR pair.
__device__ inline void async_copy64B(const _Float16* g, _Float16* lds_generic) {
    unsigned lo = (unsigned)(uintptr_t)lds_generic;
    unsigned long long ga = (unsigned long long)(uintptr_t)g;
    asm volatile(
        "global_load_async_to_lds_b128 %0, %1, off\n\t"
        "global_load_async_to_lds_b128 %0, %1, off offset:16\n\t"
        "global_load_async_to_lds_b128 %0, %1, off offset:32\n\t"
        "global_load_async_to_lds_b128 %0, %1, off offset:48"
        :: "v"(lo), "v"(ga) : "memory");
}
__device__ inline void wait_async_le8() { asm volatile("s_wait_asynccnt 0x8" ::: "memory"); }
__device__ inline void wait_async_0()   { asm volatile("s_wait_asynccnt 0x0" ::: "memory"); }

// ---------------- weight conversion ----------------

__global__ void __launch_bounds__(256) k_convert(const float* __restrict__ s,
                                                 _Float16* __restrict__ d, int n) {
    int i = blockIdx.x * 256 + threadIdx.x;
    if (i < n) d[i] = (_Float16)s[i];
}

__global__ void __launch_bounds__(256) k_convert_qkvw(const float* __restrict__ s,
                                                      _Float16* __restrict__ d, int n) {
    int i = blockIdx.x * 256 + threadIdx.x;
    if (i < n) {
        int row = i / DIM;                 // [2304 x 768], rows <768 are Q
        float sc = (row < DIM) ? SCALE : 1.0f;
        d[i] = (_Float16)(s[i] * sc);
    }
}

__global__ void __launch_bounds__(256) k_build_qkvb(const float* __restrict__ qb,
                                                    const float* __restrict__ vb,
                                                    float* __restrict__ d) {
    int i = blockIdx.x * 256 + threadIdx.x;
    if (i < 3 * DIM) {
        float v = 0.0f;
        if (i < DIM) v = qb[i] * SCALE;
        else if (i >= 2 * DIM) v = vb[i - 2 * DIM];
        d[i] = v;
    }
}

// ---------------- LayerNorm kernels ----------------

__device__ inline void ln_row(const float* __restrict__ xr,
                              const float* __restrict__ w, const float* __restrict__ bp,
                              _Float16* __restrict__ dst, float* red) {
    const int tid = threadIdx.x;
    float v[3], s = 0.f, s2 = 0.f;
#pragma unroll
    for (int i = 0; i < 3; ++i) { v[i] = xr[tid + i * 256]; s += v[i]; s2 += v[i] * v[i]; }
    red[tid] = s; __syncthreads();
    for (int off = 128; off > 0; off >>= 1) { if (tid < off) red[tid] += red[tid + off]; __syncthreads(); }
    float mu = red[0] * (1.0f / 768.0f); __syncthreads();
    red[tid] = s2; __syncthreads();
    for (int off = 128; off > 0; off >>= 1) { if (tid < off) red[tid] += red[tid + off]; __syncthreads(); }
    float var = red[0] * (1.0f / 768.0f) - mu * mu;
    float rstd = rsqrtf(var + 1e-6f);
#pragma unroll
    for (int i = 0; i < 3; ++i) {
        int c = tid + i * 256;
        dst[c] = (_Float16)((v[i] - mu) * rstd * w[c] + bp[c]);
    }
}

__global__ void __launch_bounds__(256)
k_ln1(const float* __restrict__ x, const float* __restrict__ w,
      const float* __restrict__ b, _Float16* __restrict__ out) {
    __shared__ float red[256];
    size_t row = blockIdx.x;
    ln_row(x + row * DIM, w, b, out + row * DIM, red);
}

// LN2: route text tokens (t<40) / image tokens to compact buffers with their own params.
__global__ void __launch_bounds__(256)
k_ln2(const float* __restrict__ x1,
      const float* __restrict__ wt, const float* __restrict__ bt,
      const float* __restrict__ wi, const float* __restrict__ bi,
      _Float16* __restrict__ ht, _Float16* __restrict__ hi) {
    __shared__ float red[256];
    int row = blockIdx.x;
    int t = row % NN, bidx = row / NN;
    bool is_text = (t < TEXT_LEN);
    const float* w  = is_text ? wt : wi;
    const float* bp = is_text ? bt : bi;
    _Float16* dst = is_text ? (ht + (size_t)(bidx * TEXT_LEN + t) * DIM)
                            : (hi + (size_t)(bidx * IMAG_LEN + (t - TEXT_LEN)) * DIM);
    ln_row(x1 + (size_t)row * DIM, w, bp, dst, red);
}

// ---------------- generic tiled WMMA GEMM: C[M,N] = A[M,K] * W[N,K]^T + bias ----------------
// 128 threads / 4 waves, 64x64 C tile, K-step 64, double-buffered async-to-LDS staging.
// MODE 0: store f16 (QKV)
// MODE 1: exact GELU, store f16 (FC1)
// MODE 2: outf[m] = res[m] + gamma[n]*val   (proj residual, N==768)
// MODE 3: like 2 but scatter compact row m -> global row b*236 + tok_base + m%rows_per_b (FC2)

template <int MODE>
__global__ void __launch_bounds__(128)
k_gemm(const _Float16* __restrict__ A, const _Float16* __restrict__ W,
       const float* __restrict__ bias, int M, int N, int K,
       const float* __restrict__ res, const float* __restrict__ gamma,
       float* __restrict__ outf, _Float16* __restrict__ outh,
       int rows_per_b, int tok_base) {
    __shared__ __align__(16) _Float16 As[2][64][64];
    __shared__ __align__(16) _Float16 Bs[2][64][64];
    const int tid = threadIdx.x, lane = tid & 31, wave = tid >> 5;
    const int m0 = blockIdx.y * 64, n0 = blockIdx.x * 64;
    const int lrow = tid >> 1, lhalf = (tid & 1) * 32;   // 32 f16 = 64 B per thread per tile

    v8f acc[4] = {{}, {}, {}, {}};
    const int nk = K >> 6;

    // prologue: issue tile 0 into buffer 0 (8 async b128 per wave)
    async_copy64B(A + (size_t)(m0 + lrow) * K + lhalf, &As[0][lrow][lhalf]);
    async_copy64B(W + (size_t)(n0 + lrow) * K + lhalf, &Bs[0][lrow][lhalf]);

    for (int kt = 0; kt < nk; ++kt) {
        const int cur = kt & 1;
        if (kt + 1 < nk) {  // issue next tile into other buffer (safe: barrier ended last read)
            const int kb = (kt + 1) << 6;
            async_copy64B(A + (size_t)(m0 + lrow) * K + kb + lhalf, &As[1 - cur][lrow][lhalf]);
            async_copy64B(W + (size_t)(n0 + lrow) * K + kb + lhalf, &Bs[1 - cur][lrow][lhalf]);
            wait_async_le8();          // current tile's 8 copies complete (in-order)
        } else {
            wait_async_0();
        }
        __syncthreads();               // all waves' data visible

        const _Float16* arow = &As[cur][wave * 16 + (lane & 15)][0];
        v16h a0 = frag_a(arow, 0, lane);
        v16h a1 = frag_a(arow, 32, lane);
#pragma unroll
        for (int nt = 0; nt < 4; ++nt) {
            const _Float16* bcol = &Bs[cur][nt * 16 + (lane & 15)][0];
            v16h b0 = frag_b(bcol, 0, lane);
            v16h b1 = frag_b(bcol, 32, lane);
            acc[nt] = wmma16(a0, b0, acc[nt]);
            acc[nt] = wmma16(a1, b1, acc[nt]);
        }
        __syncthreads();               // done reading cur before it is overwritten
    }

#pragma unroll
    for (int nt = 0; nt < 4; ++nt) {
        int n = n0 + nt * 16 + (lane & 15);
        float bv = bias[n];
#pragma unroll
        for (int r = 0; r < 8; ++r) {
            int m = m0 + wave * 16 + r + ((lane & 16) >> 1);
            float v = acc[nt][r] + bv;
            if (MODE == 0) {
                outh[(size_t)m * N + n] = (_Float16)v;
            } else if (MODE == 1) {
                float g = 0.5f * v * (1.0f + erff(v * 0.70710678f));
                outh[(size_t)m * N + n] = (_Float16)g;
            } else if (MODE == 2) {
                size_t o = (size_t)m * DIM + n;
                outf[o] = res[o] + gamma[n] * v;
            } else {
                int bb = m / rows_per_b;
                int t  = tok_base + m % rows_per_b;
                size_t o = (size_t)(bb * NN + t) * DIM + n;
                outf[o] = res[o] + gamma[n] * v;
            }
        }
    }
}

// ---------------- attention: one block per (b, head) ----------------
// qkv layout: [3776, 2304] f16, q pre-scaled. Tokens padded to 256 (clamped loads; padded
// columns skipped by softmax, P set to 0, padded rows never stored).

__global__ void __launch_bounds__(128)
k_attn(const _Float16* __restrict__ qkv, const float* __restrict__ relbias,
       const unsigned char* __restrict__ mask, _Float16* __restrict__ attn_out) {
    extern __shared__ __align__(16) char smem[];
    _Float16* ks = (_Float16*)smem;                   // [256][64]  keys, [token][d]
    _Float16* vT = ks + 256 * 64;                     // [64][256]  values transposed [d][token]
    float*    Sw = (float*)(vT + 64 * 256);           // per-wave [16][256] scores
    _Float16* Pw = (_Float16*)(Sw + 4 * 16 * 256);    // per-wave [16][256] probs f16

    const int tid = threadIdx.x, lane = tid & 31, wave = tid >> 5;
    const int b = blockIdx.x / HEADS, h = blockIdx.x % HEADS;

    // stage K and V^T
    for (int t = tid; t < 256; t += 128) {
        int tc = t < NN ? t : NN - 1;
        const size_t rbase = (size_t)(b * NN + tc) * (3 * DIM);
        const uint4* sk = (const uint4*)(qkv + rbase + DIM + h * HEAD_DIM);
        uint4* dk = (uint4*)(ks + t * 64);
#pragma unroll
        for (int i = 0; i < 8; ++i) dk[i] = sk[i];
        const uint4* sv = (const uint4*)(qkv + rbase + 2 * DIM + h * HEAD_DIM);
        __align__(16) _Float16 tmp[64];
        uint4* tp = (uint4*)tmp;
#pragma unroll
        for (int i = 0; i < 8; ++i) tp[i] = sv[i];
        for (int d = 0; d < 64; ++d) vT[d * 256 + t] = tmp[d];
    }
    __syncthreads();

    float*    Sm = Sw + wave * 16 * 256;
    _Float16* Pm = Pw + wave * 16 * 256;

    for (int tile = wave; tile < 15; tile += 4) {            // 15 row tiles cover 240>=236 rows
        int trow = tile * 16 + (lane & 15);
        int tc = trow < NN ? trow : NN - 1;
        const _Float16* qrow = qkv + (size_t)(b * NN + tc) * (3 * DIM) + h * HEAD_DIM;
        v16h a0 = frag_a(qrow, 0, lane);
        v16h a1 = frag_a(qrow, 32, lane);

        // S = q k^T  (16 x 256)
        for (int j = 0; j < 16; ++j) {
            const _Float16* kcol = ks + (j * 16 + (lane & 15)) * 64;
            v8f s = {};
            s = wmma16(a0, frag_b(kcol, 0, lane), s);
            s = wmma16(a1, frag_b(kcol, 32, lane), s);
#pragma unroll
            for (int r = 0; r < 8; ++r)
                Sm[(r + ((lane & 16) >> 1)) * 256 + j * 16 + (lane & 15)] = s[r];
        }

        // softmax with rel-pos bias + mask (lanes 0-15, one row each; DS ops in-order per wave)
        if (lane < 16) {
            int r = lane, tg = tile * 16 + r;
            if (tg < NN) {
                float* Srow = Sm + r * 256;
                const float* brow = relbias + ((size_t)h * NN + tg) * NN;
                const unsigned char* mrow = mask + b * NN;
                float mx = -3.0e38f;
                for (int c = 0; c < NN; ++c) {
                    float v = Srow[c] + brow[c];
                    if (!mrow[c]) v = -3.0e38f;
                    Srow[c] = v;
                    mx = v > mx ? v : mx;
                }
                float sum = 0.f;
                for (int c = 0; c < NN; ++c) { float e = __expf(Srow[c] - mx); Srow[c] = e; sum += e; }
                float rs = sum > 0.f ? 1.0f / sum : 0.0f;
                for (int c = 0; c < NN; ++c)  Pm[r * 256 + c] = (_Float16)(Srow[c] * rs);
                for (int c = NN; c < 256; ++c) Pm[r * 256 + c] = (_Float16)0.0f;
            } else {
                for (int c = 0; c < 256; ++c) Pm[r * 256 + c] = (_Float16)0.0f;
            }
        }

        // O = P V  (16 x 64)
        v8f o[4] = {{}, {}, {}, {}};
        for (int s8 = 0; s8 < 8; ++s8) {
            v16h ap = frag_a(Pm + (lane & 15) * 256, s8 * 32, lane);
#pragma unroll
            for (int nt = 0; nt < 4; ++nt) {
                v16h bv = frag_b(vT + (nt * 16 + (lane & 15)) * 256, s8 * 32, lane);
                o[nt] = wmma16(ap, bv, o[nt]);
            }
        }
#pragma unroll
        for (int nt = 0; nt < 4; ++nt)
#pragma unroll
            for (int r = 0; r < 8; ++r) {
                int tg = tile * 16 + r + ((lane & 16) >> 1);
                if (tg < NN)
                    attn_out[(size_t)(b * NN + tg) * DIM + h * HEAD_DIM + nt * 16 + (lane & 15)] =
                        (_Float16)o[nt][r];
            }
    }
}

// ---------------- orchestration ----------------

static inline size_t align256(size_t x) { return (x + 255) & ~(size_t)255; }

extern "C" void kernel_launch(void* const* d_in, const int* in_sizes, int n_in,
                              void* d_out, int out_size, void* d_ws, size_t ws_size,
                              hipStream_t stream) {
    const float* x        = (const float*)d_in[0];
    const unsigned char* mask = (const unsigned char*)d_in[1];   // jnp bool_ -> 1 byte
    const float* relbias  = (const float*)d_in[2];
    const float* qkv_w    = (const float*)d_in[3];
    const float* q_bias   = (const float*)d_in[4];
    const float* v_bias   = (const float*)d_in[5];
    const float* proj_w   = (const float*)d_in[6];
    const float* proj_b   = (const float*)d_in[7];
    const float* norm1_w  = (const float*)d_in[8];
    const float* norm1_b  = (const float*)d_in[9];
    const float* n2t_w    = (const float*)d_in[10];
    const float* n2t_b    = (const float*)d_in[11];
    const float* n2i_w    = (const float*)d_in[12];
    const float* n2i_b    = (const float*)d_in[13];
    const float* fc1t_w   = (const float*)d_in[14];
    const float* fc1t_b   = (const float*)d_in[15];
    const float* fc2t_w   = (const float*)d_in[16];
    const float* fc2t_b   = (const float*)d_in[17];
    const float* fc1i_w   = (const float*)d_in[18];
    const float* fc1i_b   = (const float*)d_in[19];
    const float* fc2i_w   = (const float*)d_in[20];
    const float* fc2i_b   = (const float*)d_in[21];
    const float* gamma1   = (const float*)d_in[22];
    const float* gamma2   = (const float*)d_in[23];
    float* out = (float*)d_out;

    // workspace carve
    char* p = (char*)d_ws;
    size_t off = 0;
    auto carve = [&](size_t bytes) { void* r = p + off; off = align256(off + bytes); return r; };
    _Float16* h16     = (_Float16*)carve((size_t)ROWS * DIM * 2);
    _Float16* qkvW16  = (_Float16*)carve((size_t)3 * DIM * DIM * 2);
    float*    qkvB    = (float*)   carve((size_t)3 * DIM * 4);
    _Float16* qkv16   = (_Float16*)carve((size_t)ROWS * 3 * DIM * 2);
    _Float16* attnO16 = (_Float16*)carve((size_t)ROWS * DIM * 2);
    _Float16* projW16 = (_Float16*)carve((size_t)DIM * DIM * 2);
    float*    x1      = (float*)   carve((size_t)ROWS * DIM * 4);
    _Float16* h2t     = (_Float16*)carve((size_t)BB * TEXT_LEN * DIM * 2);
    _Float16* h2i     = (_Float16*)carve((size_t)BB * IMAG_LEN * DIM * 2);
    _Float16* fc1t16  = (_Float16*)carve((size_t)HIDDEN * DIM * 2);
    _Float16* fc1i16  = (_Float16*)carve((size_t)HIDDEN * DIM * 2);
    _Float16* fc2t16  = (_Float16*)carve((size_t)DIM * HIDDEN * 2);
    _Float16* fc2i16  = (_Float16*)carve((size_t)DIM * HIDDEN * 2);
    _Float16* gt16    = (_Float16*)carve((size_t)BB * TEXT_LEN * HIDDEN * 2);
    _Float16* gi16    = (_Float16*)carve((size_t)BB * IMAG_LEN * HIDDEN * 2);
    (void)ws_size; (void)n_in; (void)in_sizes; (void)out_size;

    auto blocks = [](int n) { return (n + 255) / 256; };

    // 1) weight conversion
    k_convert_qkvw<<<blocks(3 * DIM * DIM), 256, 0, stream>>>(qkv_w, qkvW16, 3 * DIM * DIM);
    k_build_qkvb<<<blocks(3 * DIM), 256, 0, stream>>>(q_bias, v_bias, qkvB);
    k_convert<<<blocks(DIM * DIM), 256, 0, stream>>>(proj_w, projW16, DIM * DIM);
    k_convert<<<blocks(HIDDEN * DIM), 256, 0, stream>>>(fc1t_w, fc1t16, HIDDEN * DIM);
    k_convert<<<blocks(HIDDEN * DIM), 256, 0, stream>>>(fc1i_w, fc1i16, HIDDEN * DIM);
    k_convert<<<blocks(DIM * HIDDEN), 256, 0, stream>>>(fc2t_w, fc2t16, DIM * HIDDEN);
    k_convert<<<blocks(DIM * HIDDEN), 256, 0, stream>>>(fc2i_w, fc2i16, DIM * HIDDEN);

    // 2) LN1
    k_ln1<<<ROWS, 256, 0, stream>>>(x, norm1_w, norm1_b, h16);

    // 3) QKV GEMM: [3776,768] x [2304,768]^T -> f16 (q pre-scaled)
    k_gemm<0><<<dim3(3 * DIM / 64, ROWS / 64), 128, 0, stream>>>(
        h16, qkvW16, qkvB, ROWS, 3 * DIM, DIM, nullptr, nullptr, nullptr, qkv16, 0, 0);

    // 4) attention (dynamic LDS: K + V^T + per-wave S/P = 160 KB, legal on CDNA5 WGP)
    size_t attn_lds = (size_t)256 * 64 * 2 + (size_t)64 * 256 * 2 +
                      (size_t)4 * 16 * 256 * 4 + (size_t)4 * 16 * 256 * 2;
    k_attn<<<BB * HEADS, 128, attn_lds, stream>>>(qkv16, relbias, mask, attnO16);

    // 5) proj + residual: x1 = x + gamma1 * (attnO @ projW^T + proj_b)
    k_gemm<2><<<dim3(DIM / 64, ROWS / 64), 128, 0, stream>>>(
        attnO16, projW16, proj_b, ROWS, DIM, DIM, x, gamma1, x1, nullptr, 0, 0);

    // 6) LN2 (split text / image into compact buffers)
    k_ln2<<<ROWS, 256, 0, stream>>>(x1, n2t_w, n2t_b, n2i_w, n2i_b, h2t, h2i);

    // 7) FC1 + exact GELU
    k_gemm<1><<<dim3(HIDDEN / 64, BB * TEXT_LEN / 64), 128, 0, stream>>>(
        h2t, fc1t16, fc1t_b, BB * TEXT_LEN, HIDDEN, DIM, nullptr, nullptr, nullptr, gt16, 0, 0);
    k_gemm<1><<<dim3(HIDDEN / 64, BB * IMAG_LEN / 64), 128, 0, stream>>>(
        h2i, fc1i16, fc1i_b, BB * IMAG_LEN, HIDDEN, DIM, nullptr, nullptr, nullptr, gi16, 0, 0);

    // 8) FC2 + gamma2-residual, scattered back to original token order = output order
    k_gemm<3><<<dim3(DIM / 64, BB * TEXT_LEN / 64), 128, 0, stream>>>(
        gt16, fc2t16, fc2t_b, BB * TEXT_LEN, DIM, HIDDEN, x1, gamma2, out, nullptr, TEXT_LEN, 0);
    k_gemm<3><<<dim3(DIM / 64, BB * IMAG_LEN / 64), 128, 0, stream>>>(
        gi16, fc2i16, fc2i_b, BB * IMAG_LEN, DIM, HIDDEN, x1, gamma2, out, nullptr, IMAG_LEN, TEXT_LEN);
}
